// GraphEncoder_67259187855555
// MI455X (gfx1250) — matrix operand order repaired and checked by reading, not compiled, not run on hardware
//
#include <hip/hip_runtime.h>

// ---------------------------------------------------------------------------
// GCN 3-layer encoder for MI455X (gfx1250, wave32).
//   per layer: H = X @ W  (WMMA bf16->f32),  agg = scatter_add over edges,
//              act = relu(agg + H*dinv^2 + b)
//   final: column mean.
// Aggregation working set (~128MB) fits MI455X's 192MB L2, so the edge
// gather/atomic-scatter runs at L2 speed; GEMM uses v_wmma_f32_16x16x32_bf16
// with W staged in LDS (bf16, k-contiguous) and b128 A loads.
// ---------------------------------------------------------------------------

typedef __attribute__((ext_vector_type(16))) __bf16 v16bf;
typedef __attribute__((ext_vector_type(8)))  float  v8f;

#define D_FEAT 128
#define WT_STRIDE 136   // 128 + 8 pad: breaks 256B-stride LDS bank conflicts

// ---------------------------------------------------------------------------
// Generic float zero (grid-stride).
// ---------------------------------------------------------------------------
__global__ void zero_f32(float* __restrict__ p, int n) {
    int i = blockIdx.x * blockDim.x + threadIdx.x;
    int stride = gridDim.x * blockDim.x;
    for (; i < n; i += stride) p[i] = 0.0f;
}

// ---------------------------------------------------------------------------
// Degree: deg[dst] += 1 per edge (deg buffer pre-zeroed).
// ---------------------------------------------------------------------------
__global__ void deg_count(const long long* __restrict__ ei, float* __restrict__ deg, int E) {
    int e = blockIdx.x * blockDim.x + threadIdx.x;
    if (e < E) atomicAdd(&deg[(int)ei[(size_t)E + e]], 1.0f);
}

// dinv[i] = (deg[i] + 1)^-1/2, in place.
__global__ void deg_to_dinv(float* __restrict__ d, int n) {
    int i = blockIdx.x * blockDim.x + threadIdx.x;
    if (i < n) d[i] = rsqrtf(d[i] + 1.0f);
}

// ---------------------------------------------------------------------------
// H[N,128] = X[N,128] @ W[128,128] via V_WMMA_F32_16X16X32_BF16.
// Block = 8 waves; wave w owns column-tile w of one 16-row stripe.
// Step 1: cooperatively stage W into LDS as bf16, transposed to [n][k]
//         (k-contiguous) so each lane's 16 B-operand K-values are two
//         ds_load_b128s.
// Step 2: per wave, 4 WMMAs over K=128.
// VGPR layouts per CDNA5 ISA 7.12.2:
//   A (16x32 bf16): lane half 0 -> K 0..7 & 16..23, half 1 -> K 8..15 & 24..31
//   B (32x16 bf16): lane half 0 -> K 0..7 & 16..23, half 1 -> K 8..15 & 24..31
//   C/D (16x16 f32): VGPR r -> row r (+8 for lane half 1), col = lane & 15
// ---------------------------------------------------------------------------
__global__ void __launch_bounds__(256)
gemm_wmma_bf16(const float* __restrict__ X,
               const float* __restrict__ W,
               float* __restrict__ H, int Nrows) {
    __shared__ __bf16 Wt[D_FEAT * WT_STRIDE];          // ~34 KB

    const int lane = threadIdx.x;          // 0..31 (wave32)
    const int half = lane >> 4;            // 0 | 1
    const int lm   = lane & 15;
    const int tm   = blockIdx.x;           // row-tile
    const int tn   = threadIdx.y;          // col-tile 0..7
    const int tid  = tn * 32 + lane;

    // --- stage W -> LDS: coalesced float4 reads, bf16 transposed writes ---
    for (int q = tid; q < D_FEAT * (D_FEAT / 4); q += 256) {
        const int k  = q >> 5;             // row of W (fan_in index)
        const int nq = q & 31;             // float4 index along n
        const float4 w4 = ((const float4*)(W + (size_t)k * D_FEAT))[nq];
        const int n0 = nq * 4;
        Wt[(n0 + 0) * WT_STRIDE + k] = (__bf16)w4.x;
        Wt[(n0 + 1) * WT_STRIDE + k] = (__bf16)w4.y;
        Wt[(n0 + 2) * WT_STRIDE + k] = (__bf16)w4.z;
        Wt[(n0 + 3) * WT_STRIDE + k] = (__bf16)w4.w;
    }
    __syncthreads();

    int rowA = tm * 16 + lm;
    int rowA_ld = rowA < Nrows ? rowA : Nrows - 1;     // clamp loads, tail tile
    const int colB = tn * 16 + lm;

    v8f acc = {};
    #pragma unroll
    for (int k0 = 0; k0 < D_FEAT; k0 += 32) {
        // A operand: four b128 loads of this lane's K-groups
        const float4* xv = (const float4*)(X + (size_t)rowA_ld * D_FEAT + k0);
        const float4 p0 = xv[half * 2 + 0];            // K = k0 + half*8 + 0..3
        const float4 p1 = xv[half * 2 + 1];            //               + 4..7
        const float4 p2 = xv[half * 2 + 4];            // K = k0+16+half*8 +0..3
        const float4 p3 = xv[half * 2 + 5];            //                  +4..7
        v16bf a;
        a[0]  = (__bf16)p0.x; a[1]  = (__bf16)p0.y; a[2]  = (__bf16)p0.z; a[3]  = (__bf16)p0.w;
        a[4]  = (__bf16)p1.x; a[5]  = (__bf16)p1.y; a[6]  = (__bf16)p1.z; a[7]  = (__bf16)p1.w;
        a[8]  = (__bf16)p2.x; a[9]  = (__bf16)p2.y; a[10] = (__bf16)p2.z; a[11] = (__bf16)p2.w;
        a[12] = (__bf16)p3.x; a[13] = (__bf16)p3.y; a[14] = (__bf16)p3.z; a[15] = (__bf16)p3.w;

        // B operand: K = k0+half*8+0..7 and k0+16+half*8+0..7 from LDS (16B-aligned)
        const uint4* bls = (const uint4*)&Wt[(size_t)colB * WT_STRIDE + k0 + half * 8];
        union { struct { uint4 lo, hi; } q; v16bf v; } ub;
        ub.q.lo = bls[0];
        ub.q.hi = bls[2];

        acc = __builtin_amdgcn_wmma_f32_16x16x32_bf16(
            /*neg_a=*/false, a, /*neg_b=*/false, ub.v,
            /*c_mod=*/(short)0, acc, /*reuse_a=*/false, /*reuse_b=*/false);
    }

    #pragma unroll
    for (int r = 0; r < 8; ++r) {
        const int row = tm * 16 + r + half * 8;
        if (row < Nrows)
            H[(size_t)row * D_FEAT + tn * 16 + lm] = acc[r];
    }
}

// ---------------------------------------------------------------------------
// Edge scatter: one wave32 per edge; lane handles 4 consecutive features.
//   agg[dst] += h[src] * dinv[src]*dinv[dst]
// ---------------------------------------------------------------------------
__global__ void edge_scatter(const long long* __restrict__ ei,
                             const float* __restrict__ H,
                             const float* __restrict__ dinv,
                             float* __restrict__ agg, int E) {
    int wave = (int)((blockIdx.x * (size_t)blockDim.x + threadIdx.x) >> 5);
    int lane = threadIdx.x & 31;
    if (wave >= E) return;

    const int s = (int)ei[wave];
    const int d = (int)ei[(size_t)E + wave];
    const float nrm = dinv[s] * dinv[d];

    const float4 v = ((const float4*)(H + (size_t)s * D_FEAT))[lane];
    float* ad = agg + (size_t)d * D_FEAT + lane * 4;
    atomicAdd(ad + 0, v.x * nrm);
    atomicAdd(ad + 1, v.y * nrm);
    atomicAdd(ad + 2, v.z * nrm);
    atomicAdd(ad + 3, v.w * nrm);
}

// ---------------------------------------------------------------------------
// act = relu(agg + H*dinv^2 + bias), written in place into agg.
// ---------------------------------------------------------------------------
__global__ void finish_layer(const float* __restrict__ H,
                             const float* __restrict__ dinv,
                             const float* __restrict__ bias,
                             float* __restrict__ agg, int total) {
    int i = blockIdx.x * blockDim.x + threadIdx.x;
    if (i >= total) return;
    const int row = i >> 7;          // /128
    const int col = i & (D_FEAT - 1);
    const float di = dinv[row];
    const float v = agg[i] + H[i] * di * di + bias[col];
    agg[i] = fmaxf(v, 0.0f);
}

// ---------------------------------------------------------------------------
// out[col] = mean over rows (d_out pre-zeroed). 128 threads per block = one
// column each; each block sums a strided slice of rows, one atomic per column.
// ---------------------------------------------------------------------------
__global__ void col_mean(const float* __restrict__ act, float* __restrict__ out,
                         int Nrows) {
    const int col = threadIdx.x;       // 0..127
    float s = 0.0f;
    for (int r = blockIdx.x; r < Nrows; r += gridDim.x)
        s += act[(size_t)r * D_FEAT + col];
    atomicAdd(&out[col], s * (1.0f / (float)Nrows));
}

// ---------------------------------------------------------------------------
// Host-side orchestration (graph-capture safe: all launches on `stream`).
// ---------------------------------------------------------------------------
extern "C" void kernel_launch(void* const* d_in, const int* in_sizes, int n_in,
                              void* d_out, int out_size, void* d_ws, size_t ws_size,
                              hipStream_t stream) {
    const float*     x  = (const float*)d_in[0];
    const long long* ei = (const long long*)d_in[1];   // int64 [2, E]
    const float* Wl[3] = { (const float*)d_in[2], (const float*)d_in[4], (const float*)d_in[6] };
    const float* bl[3] = { (const float*)d_in[3], (const float*)d_in[5], (const float*)d_in[7] };

    const int N = in_sizes[0] / D_FEAT;     // 100000
    const int E = in_sizes[1] / 2;          // 1600000
    const int NF = N * D_FEAT;

    // Workspace layout (floats): [dinv: N][bufH: N*128][bufAgg: N*128]
    float* dinv   = (float*)d_ws;
    float* bufH   = dinv + N;
    float* bufAgg = bufH + (size_t)NF;
    float* outF   = (float*)d_out;

    const int ZB = 256;

    // --- degree -> dinv (once; shared by all 3 layers) ---
    zero_f32<<<(N + ZB - 1) / ZB, ZB, 0, stream>>>(dinv, N);
    deg_count<<<(E + ZB - 1) / ZB, ZB, 0, stream>>>(ei, dinv, E);
    deg_to_dinv<<<(N + ZB - 1) / ZB, ZB, 0, stream>>>(dinv, N);

    const dim3 gemmGrid((N + 15) / 16, 1, 1);
    const dim3 gemmBlock(32, 8, 1);                     // 8 waves = 8 col-tiles
    const int scatterBlocks = (int)(((size_t)E * 32 + ZB - 1) / ZB);

    const float* actIn = x;
    for (int L = 0; L < 3; ++L) {
        gemm_wmma_bf16<<<gemmGrid, gemmBlock, 0, stream>>>(actIn, Wl[L], bufH, N);
        zero_f32<<<4096, ZB, 0, stream>>>(bufAgg, NF);
        edge_scatter<<<scatterBlocks, ZB, 0, stream>>>(ei, bufH, dinv, bufAgg, E);
        finish_layer<<<(NF + ZB - 1) / ZB, ZB, 0, stream>>>(bufH, dinv, bl[L], bufAgg, NF);
        actIn = bufAgg;                                  // next layer input
    }

    zero_f32<<<1, D_FEAT, 0, stream>>>(outF, out_size);
    col_mean<<<1024, D_FEAT, 0, stream>>>(bufAgg, outF, N);
}
